// SupervisedGraphsage_84963043049899
// MI455X (gfx1250) — matrix operand order repaired and checked by reading, not compiled, not run on hardware
//
#include <hip/hip_runtime.h>

typedef __attribute__((ext_vector_type(16))) __bf16 v16bf;
typedef __attribute__((ext_vector_type(8)))  __bf16 v8bf;
typedef __attribute__((ext_vector_type(4)))  __bf16 v4bf;
typedef __attribute__((ext_vector_type(8)))  float  v8f;

#define D_FEAT 512
#define HID2   512   // 2*HID
#define BSZ    512
#define S1N    25
#define S2N    10
#define NCLS   50

// ---------------------------------------------------------------------------
// Weight convert: W[k][n] f32 (K x N, N=256, K=512) -> Wt[n][k] bf16 (N x K)
// ---------------------------------------------------------------------------
__global__ __launch_bounds__(256) void wt_transpose_bf16(
    const float* __restrict__ w, __bf16* __restrict__ wt, int K, int N) {
  int i = blockIdx.x * blockDim.x + threadIdx.x;
  if (i < K * N) {
    int k = i / N, n = i % N;
    wt[(size_t)n * K + k] = (__bf16)w[i];
  }
}

// ---------------------------------------------------------------------------
// Fused gather + mean over S sampled neighbor rows, f32 accumulate -> bf16.
// grid = G groups, 128 threads, each thread owns 4 consecutive columns.
// S==1 degenerates to a gather+convert.
// ---------------------------------------------------------------------------
__global__ __launch_bounds__(128) void gather_mean_bf16(
    const float* __restrict__ feat, const int* __restrict__ idx,
    __bf16* __restrict__ out, int S, float inv) {
  int g = blockIdx.x;
  int c = threadIdx.x * 4;
  float ax = 0.f, ay = 0.f, az = 0.f, aw = 0.f;
  const int base = g * S;
  for (int s = 0; s < S; ++s) {
    size_t row = (size_t)idx[base + s];
    const float4 v = *(const float4*)(feat + row * D_FEAT + c);
    ax += v.x; ay += v.y; az += v.z; aw += v.w;
  }
  v4bf o;
  o[0] = (__bf16)(ax * inv); o[1] = (__bf16)(ay * inv);
  o[2] = (__bf16)(az * inv); o[3] = (__bf16)(aw * inv);
  *(v4bf*)(out + (size_t)g * D_FEAT + c) = o;
}

// ---------------------------------------------------------------------------
// Mean over S consecutive rows of a dense f32 [G*S, 512] -> bf16 [G, 512]
// ---------------------------------------------------------------------------
__global__ __launch_bounds__(128) void mean_rows_bf16(
    const float* __restrict__ in, __bf16* __restrict__ out, int S, float inv) {
  int g = blockIdx.x;
  int c = threadIdx.x * 4;
  float ax = 0.f, ay = 0.f, az = 0.f, aw = 0.f;
  for (int s = 0; s < S; ++s) {
    const float4 v = *(const float4*)(in + (size_t)(g * S + s) * HID2 + c);
    ax += v.x; ay += v.y; az += v.z; aw += v.w;
  }
  v4bf o;
  o[0] = (__bf16)(ax * inv); o[1] = (__bf16)(ay * inv);
  o[2] = (__bf16)(az * inv); o[3] = (__bf16)(aw * inv);
  *(v4bf*)(out + (size_t)g * HID2 + c) = o;
}

__global__ __launch_bounds__(256) void f32_to_bf16(
    const float* __restrict__ in, __bf16* __restrict__ out, int n) {
  int i = blockIdx.x * blockDim.x + threadIdx.x;
  if (i < n) out[i] = (__bf16)in[i];
}

// ---------------------------------------------------------------------------
// Dual-source GEMM with v_wmma_f32_16x16x32_bf16 (wave32, one wave per
// 16x16 output tile). out[M,512]: column tiles 0..15 = act(A1 @ W1t^T),
// column tiles 16..31 = act(A2 @ W2t^T). A*: bf16 [M,512] row-major.
// W*t: bf16 [256,512] N-major (so both fragments load contiguously).
// K = 512, tiled in 16 steps of 32.
// ---------------------------------------------------------------------------
__global__ __launch_bounds__(256) void gemm_dual_wmma(
    const __bf16* __restrict__ A1, const __bf16* __restrict__ A2,
    const __bf16* __restrict__ W1t, const __bf16* __restrict__ W2t,
    float* __restrict__ out, int M, int relu) {
  const int wave = threadIdx.x >> 5;
  const int lane = threadIdx.x & 31;
  const int half = lane >> 4;    // 0: lanes 0-15, 1: lanes 16-31
  const int r    = lane & 15;

  int tile = blockIdx.x * 8 + wave;      // 8 waves per block
  int mt = tile >> 5;                    // 32 column tiles
  int nt = tile & 31;
  const int m0 = mt * 16;
  if (m0 >= M) return;                   // wave-uniform

  const __bf16* A  = (nt < 16) ? A1  : A2;
  const __bf16* Wt = (nt < 16) ? W1t : W2t;
  const int n0 = (nt & 15) * 16;

  const __bf16* arow = A  + (size_t)(m0 + r) * HID2;  // K stride 512
  const __bf16* brow = Wt + (size_t)(n0 + r) * HID2;

  v8f acc = {};
#pragma unroll 4
  for (int k0 = 0; k0 < 512; k0 += 32) {
    // A fragment (16x32 bf16): lanes 0-15 -> K k0+0..7 / k0+16..23,
    //                          lanes 16-31 -> K k0+8..15 / k0+24..31
    v8bf a_lo = *(const v8bf*)(arow + k0 + half * 8);
    v8bf a_hi = *(const v8bf*)(arow + k0 + 16 + half * 8);
    // B fragment (32x16 bf16): lanes 0-15 -> K k0+0..7 / k0+16..23,
    //                          lanes 16-31 -> K k0+8..15 / k0+24..31
    v8bf b_lo = *(const v8bf*)(brow + k0 + half * 8);
    v8bf b_hi = *(const v8bf*)(brow + k0 + 16 + half * 8);
    v16bf a, b;
#pragma unroll
    for (int j = 0; j < 8; ++j) {
      a[j] = a_lo[j]; a[j + 8] = a_hi[j];
      b[j] = b_lo[j]; b[j + 8] = b_hi[j];
    }
    acc = __builtin_amdgcn_wmma_f32_16x16x32_bf16(
        false, a, false, b, (short)0, acc, false, false);
  }

  // C/D layout: VGPR g -> row m0+g (lanes 0-15) / m0+8+g (lanes 16-31),
  // col = tile_col*16 + r
  const int col = nt * 16 + r;
#pragma unroll
  for (int g = 0; g < 8; ++g) {
    float v = acc[g];
    if (relu) v = v > 0.f ? v : 0.f;
    out[(size_t)(m0 + g + half * 8) * HID2 + col] = v;
  }
}

// ---------------------------------------------------------------------------
// Row L2-normalize (eps 1e-12) + logits = norm @ W_pred + b_pred
// one block per batch row
// ---------------------------------------------------------------------------
__global__ __launch_bounds__(256) void norm_pred(
    const float* __restrict__ outm, const float* __restrict__ Wp,
    const float* __restrict__ bp, float* __restrict__ logits) {
  __shared__ float row[HID2];
  __shared__ float red[256];
  const int b = blockIdx.x, t = threadIdx.x;
  float ss = 0.f;
  for (int c = t; c < HID2; c += 256) {
    float v = outm[(size_t)b * HID2 + c];
    row[c] = v;
    ss += v * v;
  }
  red[t] = ss;
  __syncthreads();
  for (int off = 128; off > 0; off >>= 1) {
    if (t < off) red[t] += red[t + off];
    __syncthreads();
  }
  const float inv = 1.0f / fmaxf(sqrtf(red[0]), 1e-12f);
  if (t < NCLS) {
    float acc = 0.f;
#pragma unroll 4
    for (int k = 0; k < HID2; ++k) acc += row[k] * Wp[k * NCLS + t];
    logits[b * NCLS + t] = acc * inv + bp[t];
  }
}

// ---------------------------------------------------------------------------
extern "C" void kernel_launch(void* const* d_in, const int* in_sizes, int n_in,
                              void* d_out, int out_size, void* d_ws, size_t ws_size,
                              hipStream_t stream) {
  (void)in_sizes; (void)n_in; (void)out_size; (void)ws_size;
  const float* features = (const float*)d_in[0];
  const float* W_self0  = (const float*)d_in[1];
  const float* W_neigh0 = (const float*)d_in[2];
  const float* W_self1  = (const float*)d_in[3];
  const float* W_neigh1 = (const float*)d_in[4];
  const float* W_pred   = (const float*)d_in[5];
  const float* b_pred   = (const float*)d_in[6];
  const int*   s0       = (const int*)d_in[7];
  const int*   s1       = (const int*)d_in[8];
  const int*   s2       = (const int*)d_in[9];
  float* logits = (float*)d_out;

  // workspace carve-up (256B aligned)
  char* ws = (char*)d_ws;
  size_t off = 0;
  auto carve = [&](size_t bytes) {
    char* p = ws + off;
    off = (off + bytes + 255) & ~(size_t)255;
    return p;
  };
  __bf16* ws0t   = (__bf16*)carve(512 * 256 * 2);   // W_self0^T  bf16 [256,512]
  __bf16* wn0t   = (__bf16*)carve(512 * 256 * 2);
  __bf16* ws1t   = (__bf16*)carve(512 * 256 * 2);
  __bf16* wn1t   = (__bf16*)carve(512 * 256 * 2);
  __bf16* h1bf   = (__bf16*)carve((size_t)5120 * 512 * 2);
  __bf16* mean2b = (__bf16*)carve((size_t)5120 * 512 * 2);
  __bf16* h0bf   = (__bf16*)carve((size_t)512 * 512 * 2);
  __bf16* mean1b = (__bf16*)carve((size_t)512 * 512 * 2);
  __bf16* n0bf   = (__bf16*)carve((size_t)512 * 512 * 2);
  __bf16* mn1bf  = (__bf16*)carve((size_t)512 * 512 * 2);
  float*  n1     = (float*)carve((size_t)5120 * 512 * 4);
  float*  n0     = (float*)carve((size_t)512 * 512 * 4);
  float*  outm   = (float*)carve((size_t)512 * 512 * 4);

  // 1) weights -> bf16, transposed to N-major
  wt_transpose_bf16<<<512, 256, 0, stream>>>(W_self0,  ws0t, 512, 256);
  wt_transpose_bf16<<<512, 256, 0, stream>>>(W_neigh0, wn0t, 512, 256);
  wt_transpose_bf16<<<512, 256, 0, stream>>>(W_self1,  ws1t, 512, 256);
  wt_transpose_bf16<<<512, 256, 0, stream>>>(W_neigh1, wn1t, 512, 256);

  // 2) fused gathers / neighbor means (the 262MB h2 pass never materializes h2)
  gather_mean_bf16<<<5120, 128, 0, stream>>>(features, s2, mean2b, S1N, 1.0f / S1N);
  gather_mean_bf16<<<5120, 128, 0, stream>>>(features, s1, h1bf,   1,   1.0f);
  gather_mean_bf16<<<512,  128, 0, stream>>>(features, s1, mean1b, S2N, 1.0f / S2N);
  gather_mean_bf16<<<512,  128, 0, stream>>>(features, s0, h0bf,   1,   1.0f);

  // 3) layer-0 aggregation GEMMs (relu): n1 [5120,512], n0 [512,512]
  gemm_dual_wmma<<<(5120 / 16) * 32 / 8, 256, 0, stream>>>(
      h1bf, mean2b, ws0t, wn0t, n1, 5120, 1);
  gemm_dual_wmma<<<(512 / 16) * 32 / 8, 256, 0, stream>>>(
      h0bf, mean1b, ws0t, wn0t, n0, 512, 1);

  // 4) prep layer-1 inputs
  f32_to_bf16<<<(512 * 512) / 256, 256, 0, stream>>>(n0, n0bf, 512 * 512);
  mean_rows_bf16<<<512, 128, 0, stream>>>(n1, mn1bf, S2N, 1.0f / S2N);

  // 5) layer-1 GEMM (identity act): outm [512,512]
  gemm_dual_wmma<<<(512 / 16) * 32 / 8, 256, 0, stream>>>(
      n0bf, mn1bf, ws1t, wn1t, outm, 512, 0);

  // 6) L2-normalize + predictor
  norm_pred<<<512, 256, 0, stream>>>(outm, W_pred, b_pred, logits);
}
